// LoRALinear_5506148073539
// MI455X (gfx1250) — matrix operand order repaired and checked by reading, not compiled, not run on hardware
//
#include <hip/hip_runtime.h>
#include <stdint.h>

// Problem constants (from reference setup_inputs)
#define M_TOK   8192   // B*S = 4*2048
#define N_OUT   4096   // D_out
#define K_IN    4096   // D_in
#define RANK    16
#define SCALING 2.0f   // alpha/rank = 32/16

#define BK      32     // K tile
#define SROW    40     // bf16 elems per LDS row (32 + 8 pad -> conflict-free ds_load_b128)

typedef __bf16 v4bf  __attribute__((ext_vector_type(4)));
typedef __bf16 v8bf  __attribute__((ext_vector_type(8)));
typedef __bf16 v16bf __attribute__((ext_vector_type(16)));
typedef float  v8f   __attribute__((ext_vector_type(8)));

// ---------------------------------------------------------------------------
// helpers
// ---------------------------------------------------------------------------

// gfx1250 async global->LDS copy, 16B per lane; tracked by ASYNCcnt.
// GVS mode: mem_addr = SADDR(64) + VADDR(32) + offset ; first operand = LDS byte addr VGPR.
__device__ __forceinline__ void async_ld_b128(uint32_t lds_addr, uint32_t byte_off,
                                              const void* saddr) {
  asm volatile("global_load_async_to_lds_b128 %0, %1, %2"
               :: "v"(lds_addr), "v"(byte_off), "s"(saddr)
               : "memory");
}
__device__ __forceinline__ void wait_async_le4() {
  asm volatile("s_wait_asynccnt 0x4" ::: "memory");
}
__device__ __forceinline__ void wait_async_le0() {
  asm volatile("s_wait_asynccnt 0x0" ::: "memory");
}

// low 32 bits of a SHARED-aperture generic pointer == LDS byte address
__device__ __forceinline__ uint32_t lds_addr_of(const void* p) {
  return (uint32_t)(uintptr_t)p;
}

// 16-bit A fragment 16x32 (ISA 7.12.2): lane<16 -> koff=0, lane>=16 -> koff=8;
// elems[0..7] = K[koff..koff+7], elems[8..15] = K[koff+16..koff+23]
__device__ __forceinline__ v16bf lds_fragA(const __bf16* t, int row, int koff) {
  v8bf lo = *(const v8bf*)(t + row * SROW + koff);
  v8bf hi = *(const v8bf*)(t + row * SROW + koff + 16);
  return __builtin_shufflevector(lo, hi, 0,1,2,3,4,5,6,7,8,9,10,11,12,13,14,15);
}

// 16-bit B fragment 32x16: lane holds column N=lane%16; same K order as A:
// elems[0..7] = K[boff..boff+7], elems[8..15] = K[boff+16..boff+23]  (boff = (lane>>4)*8)
__device__ __forceinline__ v16bf lds_fragB(const __bf16* t, int row, int boff) {
  v8bf lo = *(const v8bf*)(t + row * SROW + boff);
  v8bf hi = *(const v8bf*)(t + row * SROW + boff + 16);
  return __builtin_shufflevector(lo, hi, 0,1,2,3,4,5,6,7,8,9,10,11,12,13,14,15);
}

__device__ __forceinline__ v16bf pack16(float4 a, float4 b, float4 c, float4 d) {
  v16bf r;
  r[0]=(__bf16)a.x;  r[1]=(__bf16)a.y;  r[2]=(__bf16)a.z;  r[3]=(__bf16)a.w;
  r[4]=(__bf16)b.x;  r[5]=(__bf16)b.y;  r[6]=(__bf16)b.z;  r[7]=(__bf16)b.w;
  r[8]=(__bf16)c.x;  r[9]=(__bf16)c.y;  r[10]=(__bf16)c.z; r[11]=(__bf16)c.w;
  r[12]=(__bf16)d.x; r[13]=(__bf16)d.y; r[14]=(__bf16)d.z; r[15]=(__bf16)d.w;
  return r;
}

// ---------------------------------------------------------------------------
// Kernel 0: f32 -> bf16 streaming convert (8 elements / thread, exact grids)
// ---------------------------------------------------------------------------
__global__ __launch_bounds__(256)
void cvt_bf16_kernel(const float* __restrict__ s, __bf16* __restrict__ d) {
  const size_t i = ((size_t)blockIdx.x * 256 + threadIdx.x) * 8;
  float4 a = *(const float4*)(s + i);
  float4 b = *(const float4*)(s + i + 4);
  v8bf h;
  h[0]=(__bf16)a.x; h[1]=(__bf16)a.y; h[2]=(__bf16)a.z; h[3]=(__bf16)a.w;
  h[4]=(__bf16)b.x; h[5]=(__bf16)b.y; h[6]=(__bf16)b.z; h[7]=(__bf16)b.w;
  *(v8bf*)(d + i) = h;
}

// ---------------------------------------------------------------------------
// Kernel 1: XA[m][r] = sum_k x[m][k] * lora_A[r][k]   (8192x16, f32 in d_ws)
// ---------------------------------------------------------------------------
__global__ __launch_bounds__(32)
void lora_xa_kernel(const float* __restrict__ x,
                    const float* __restrict__ loraA,
                    float* __restrict__ XA) {
  const int lane = threadIdx.x & 31;
  const int l    = lane & 15;
  const int hi   = lane >> 4;
  const int rowbase = blockIdx.x * 16;

  const float* xr = x     + (size_t)(rowbase + l) * K_IN + hi * 8;   // A pattern
  const float* ar = loraA + (size_t)l * K_IN + hi * 8;               // B pattern

  v8f acc = {};
  for (int k0 = 0; k0 < K_IN; k0 += BK) {
    float4 a0 = *(const float4*)(xr + k0);
    float4 a1 = *(const float4*)(xr + k0 + 4);
    float4 a2 = *(const float4*)(xr + k0 + 16);
    float4 a3 = *(const float4*)(xr + k0 + 20);
    float4 c0 = *(const float4*)(ar + k0);
    float4 c1 = *(const float4*)(ar + k0 + 4);
    float4 c2 = *(const float4*)(ar + k0 + 16);
    float4 c3 = *(const float4*)(ar + k0 + 20);
    v16bf af = pack16(a0, a1, a2, a3);
    v16bf bf = pack16(c0, c1, c2, c3);
    acc = __builtin_amdgcn_wmma_f32_16x16x32_bf16(false, af, false, bf,
                                                  (short)0, acc, false, false);
  }
#pragma unroll
  for (int r = 0; r < 8; ++r)
    XA[(size_t)(rowbase + r + hi * 8) * RANK + l] = acc[r];
}

// ---------------------------------------------------------------------------
// Kernel 2: out = x @ W^T + b + SCALING * (XA @ loraB^T)     (bf16 inputs)
// 128x128 tile / block, 8 waves (2x4), 64x32 per wave.
// Triple-buffered LDS fed by GLOBAL_LOAD_ASYNC_TO_LDS_B128 (ASYNCcnt pipeline).
// ---------------------------------------------------------------------------
__global__ __launch_bounds__(256)
void lora_gemm_kernel(const __bf16* __restrict__ xb,
                      const __bf16* __restrict__ wb,
                      const float* __restrict__ bias,
                      const float* __restrict__ loraB,
                      const float* __restrict__ XA,
                      float* __restrict__ out) {
  __shared__ __bf16 sX[3][128 * SROW];   // 3 x 10 KB
  __shared__ __bf16 sW[3][128 * SROW];   // 3 x 10 KB

  const int tid  = threadIdx.x;
  const int lane = tid & 31;
  const int wave = tid >> 5;
  const int wm   = wave >> 2;   // 0..1 : 64-row strip
  const int wn   = wave & 3;    // 0..3 : 32-col strip
  const int l    = lane & 15;
  const int hi   = lane >> 4;

  const int blockM = blockIdx.y * 128;
  const int blockN = blockIdx.x * 128;

  // async staging: 512 16B-chunks per tile, 2 per thread per array
  const int q0 = tid;            // chunk ids
  const int q1 = tid + 256;
  const int r0 = q0 >> 2, c0 = (q0 & 3) * 8;   // row, bf16 col
  const int r1 = q1 >> 2, c1 = (q1 & 3) * 8;

  const __bf16* xsrc = xb + (size_t)blockM * K_IN;
  const __bf16* wsrc = wb + (size_t)blockN * K_IN;

  auto stage_async = [&](int buf, int k0) {
    const uint32_t gx0 = (uint32_t)((r0 * K_IN + k0 + c0) * 2);
    const uint32_t gx1 = (uint32_t)((r1 * K_IN + k0 + c1) * 2);
    async_ld_b128(lds_addr_of(&sX[buf][r0 * SROW + c0]), gx0, xsrc);
    async_ld_b128(lds_addr_of(&sX[buf][r1 * SROW + c1]), gx1, xsrc);
    async_ld_b128(lds_addr_of(&sW[buf][r0 * SROW + c0]), gx0, wsrc);
    async_ld_b128(lds_addr_of(&sW[buf][r1 * SROW + c1]), gx1, wsrc);
  };

  v8f acc[4][2] = {};

  stage_async(0, 0);

  const int NK   = K_IN / BK;   // 128
  const int koff = hi * 8;      // A-fragment half-wave K offset
  const int boff = hi * 8;      // B-fragment half-wave K offset

  int cur = 0;
  for (int kt = 0; kt < NK; ++kt) {
    const int nxt = (cur == 2) ? 0 : cur + 1;
    if (kt + 1 < NK) {
      stage_async(nxt, (kt + 1) * BK);   // 4 new copies in flight over compute
      wait_async_le4();                  // retire stage kt's copies (in-order)
    } else {
      wait_async_le0();
    }
    __syncthreads();                     // all waves' copies for buf `cur` visible

    const __bf16* xt = sX[cur];
    const __bf16* wt = sW[cur];
    v16bf b0 = lds_fragB(wt, wn * 32 + l,      boff);
    v16bf b1 = lds_fragB(wt, wn * 32 + 16 + l, boff);
#pragma unroll
    for (int i = 0; i < 4; ++i) {
      v16bf a = lds_fragA(xt, wm * 64 + i * 16 + l, koff);
      acc[i][0] = __builtin_amdgcn_wmma_f32_16x16x32_bf16(false, a, false, b0,
                                                          (short)0, acc[i][0], false, false);
      acc[i][1] = __builtin_amdgcn_wmma_f32_16x16x32_bf16(false, a, false, b1,
                                                          (short)0, acc[i][1], false, false);
    }
    cur = nxt;
  }

  // ---- Epilogue: fused bias + SCALING * (XA @ loraB^T), reuse LDS as f32 ----
  __syncthreads();
  float* xa = (float*)&sX[0][0];   // 128 x 16 f32 (8 KB)
  float* bl = (float*)&sW[0][0];   // 128 x 16 f32 (8 KB)
#pragma unroll
  for (int u = 0; u < 2; ++u) {
    const int qi = tid + u * 256;        // 0..511 float4s
    const int r  = qi >> 2;
    const int t4 = (qi & 3) * 4;
    *(float4*)&xa[r * RANK + t4] = *(const float4*)&XA[(size_t)(blockM + r) * RANK + t4];
    *(float4*)&bl[r * RANK + t4] = *(const float4*)&loraB[(size_t)(blockN + r) * RANK + t4];
  }
  __syncthreads();

  const int col0 = wn * 32 + l;
  const int col1 = col0 + 16;
  float bv0[16], bv1[16];
#pragma unroll
  for (int t = 0; t < 16; t += 4) {
    *(float4*)&bv0[t] = *(const float4*)&bl[col0 * RANK + t];
    *(float4*)&bv1[t] = *(const float4*)&bl[col1 * RANK + t];
  }
  const float bias0 = bias[blockN + col0];
  const float bias1 = bias[blockN + col1];

#pragma unroll
  for (int i = 0; i < 4; ++i) {
    const int rbase = wm * 64 + i * 16 + hi * 8;   // C/D layout: VGPR r -> M = rbase + r
#pragma unroll
    for (int r = 0; r < 8; ++r) {
      const float* xr = &xa[(rbase + r) * RANK];
      float l0 = 0.f, l1 = 0.f;
#pragma unroll
      for (int t = 0; t < 16; ++t) { l0 += xr[t] * bv0[t]; l1 += xr[t] * bv1[t]; }
      const size_t orow = (size_t)(blockM + rbase + r) * N_OUT + blockN;
      out[orow + col0] = acc[i][0][r] + bias0 + SCALING * l0;
      out[orow + col1] = acc[i][1][r] + bias1 + SCALING * l1;
    }
  }
}

// ---------------------------------------------------------------------------
extern "C" void kernel_launch(void* const* d_in, const int* in_sizes, int n_in,
                              void* d_out, int out_size, void* d_ws, size_t ws_size,
                              hipStream_t stream) {
  const float* x     = (const float*)d_in[0];   // [4,2048,4096] f32
  const float* W     = (const float*)d_in[1];   // [4096,4096]  f32
  const float* b     = (const float*)d_in[2];   // [4096]       f32
  const float* loraA = (const float*)d_in[3];   // [16,4096]    f32
  const float* loraB = (const float*)d_in[4];   // [4096,16]    f32
  float*       out   = (float*)d_out;           // [4,2048,4096] f32

  // workspace layout: bf16 x | bf16 W | XA f32
  const size_t XB_ELEMS = (size_t)M_TOK * K_IN;   // 33,554,432
  const size_t WB_ELEMS = (size_t)N_OUT * K_IN;   // 16,777,216
  __bf16* xb = (__bf16*)d_ws;
  __bf16* wbp = xb + XB_ELEMS;
  float*  XA = (float*)(wbp + WB_ELEMS);

  cvt_bf16_kernel<<<(unsigned)(XB_ELEMS / 8 / 256), 256, 0, stream>>>(x, xb);
  cvt_bf16_kernel<<<(unsigned)(WB_ELEMS / 8 / 256), 256, 0, stream>>>(W, wbp);
  lora_xa_kernel<<<M_TOK / 16, 32, 0, stream>>>(x, loraA, XA);
  lora_gemm_kernel<<<dim3(N_OUT / 128, M_TOK / 128), 256, 0, stream>>>(
      xb, wbp, b, loraB, XA, out);
}
